// Block_33998961115863
// MI455X (gfx1250) — matrix operand order repaired and checked by reading, not compiled, not run on hardware
//
#include <hip/hip_runtime.h>

typedef __attribute__((ext_vector_type(16))) _Float16 v16h;
typedef __attribute__((ext_vector_type(8)))  _Float16 v8h;
typedef __attribute__((ext_vector_type(8)))  float    v8f;

union V16H { v16h v; v8h h[2]; _Float16 e[16]; };

// ---------------- fragment loaders (CDNA5 WMMA lane layouts, ISA 7.12.2) ----------------
// A (16x32 f16, row-major [M,K]): lane<16 -> row=lane,    K = k0..k0+7  and k0+16..k0+23
//                                 lane>=16 -> row=lane-16, K = k0+8..k0+15 and k0+24..k0+31
__device__ __forceinline__ v16h load_a_frag(const _Float16* A, int lda, int m0, int k0, int lane) {
  const int row = m0 + (lane & 15);
  const int kk  = k0 + ((lane & 16) ? 8 : 0);
  const _Float16* p = A + (size_t)row * lda + kk;
  V16H r;
  r.h[0] = *(const v8h*)(p);
  r.h[1] = *(const v8h*)(p + 16);
  return r.v;
}
// B (32x16 f16) from K-contiguous source Bt[N][K]: lane -> col = n0 + lane%16,
// K = k0..k0+7 and k0+16..k0+23 (+8 for upper half-wave)
__device__ __forceinline__ v16h load_b_frag(const _Float16* Bt, int ldb, int n0, int k0, int lane) {
  const int col = n0 + (lane & 15);
  const int kk  = k0 + ((lane & 16) ? 8 : 0);
  const _Float16* p = Bt + (size_t)col * ldb + kk;
  V16H r;
  r.h[0] = *(const v8h*)(p);
  r.h[1] = *(const v8h*)(p + 16);
  return r.v;
}

// ---------------- weight convert + transpose: Wt[n*K+k] = (f16)W[k*N+n] ----------------
__global__ __launch_bounds__(256) void convert_transpose(const float* __restrict__ W,
                                                         _Float16* __restrict__ Wt,
                                                         int K, int N) {
  const size_t idx = (size_t)blockIdx.x * 256 + threadIdx.x;
  if (idx >= (size_t)K * N) return;
  const int n = (int)(idx / K);
  const int k = (int)(idx % K);
  Wt[idx] = (_Float16)W[(size_t)k * N + n];
}

// ---------------- per-head V transpose: Vt[(bh*64+d)*2048 + s] = V[b,s,h,d] ----------------
__global__ __launch_bounds__(256) void transpose_v(const _Float16* __restrict__ qkv,
                                                   _Float16* __restrict__ Vt) {
  const size_t idx = (size_t)blockIdx.x * 256 + threadIdx.x;  // 2*16*64*2048 = 4194304
  const int s  = (int)(idx & 2047);
  const int d  = (int)((idx >> 11) & 63);
  const int bh = (int)(idx >> 17);
  const int b = bh >> 4, h = bh & 15;
  Vt[idx] = qkv[((size_t)(b * 2048 + s)) * 3072 + 2048 + h * 64 + d];
}

// ---------------- layernorm (f32 in -> f16 out) ----------------
__global__ __launch_bounds__(256) void layernorm_f16(const float* __restrict__ x,
                                                     const float* __restrict__ w,
                                                     const float* __restrict__ b,
                                                     _Float16* __restrict__ out, int Hd) {
  const int row = blockIdx.x;
  const float* xr = x + (size_t)row * Hd;
  float s = 0.f, ss = 0.f;
  for (int i = threadIdx.x; i < Hd; i += 256) { float v = xr[i]; s += v; ss += v * v; }
  __shared__ float rs[256], rss[256];
  rs[threadIdx.x] = s; rss[threadIdx.x] = ss;
  __syncthreads();
  for (int o = 128; o > 0; o >>= 1) {
    if ((int)threadIdx.x < o) { rs[threadIdx.x] += rs[threadIdx.x + o]; rss[threadIdx.x] += rss[threadIdx.x + o]; }
    __syncthreads();
  }
  const float mu  = rs[0] / Hd;
  const float var = rss[0] / Hd - mu * mu;
  const float inv = rsqrtf(var + 1e-5f);
  for (int i = threadIdx.x; i < Hd; i += 256)
    out[(size_t)row * Hd + i] = (_Float16)((xr[i] - mu) * inv * w[i] + b[i]);
}

// ---------------- WMMA GEMM: C[M,N] = A[M,K](f16) * Bt[N,K](f16)^T + bias ----------------
// Block: 128 threads = 4 waves (2x2), block tile 64x128, wave tile 32x64 (8 accumulators).
// EPI 0: f16 out (bias)   EPI 1: f16 out (bias+exact GELU)   EPI 2: f32 out (bias+residual)
template <int EPI>
__global__ __launch_bounds__(128) void gemm_wmma(const _Float16* __restrict__ A,
                                                 const _Float16* __restrict__ Bt,
                                                 const float* __restrict__ bias,
                                                 const float* __restrict__ resid,
                                                 _Float16* __restrict__ outh,
                                                 float* __restrict__ outf,
                                                 int Ndim, int Kdim) {
  const int lane = threadIdx.x & 31;
  const int wave = threadIdx.x >> 5;
  const int m0 = blockIdx.y * 64 + (wave >> 1) * 32;
  const int n0 = blockIdx.x * 128 + (wave & 1) * 64;

  v8f acc[2][4] = {};

  for (int k = 0; k < Kdim; k += 32) {
    if (k + 32 < Kdim) {  // prefetch next k-tile -> global_prefetch_b8
      __builtin_prefetch(A + (size_t)(m0 + (lane & 15)) * Kdim + k + 32, 0, 3);
      __builtin_prefetch(A + (size_t)(m0 + 16 + (lane & 15)) * Kdim + k + 32, 0, 3);
      __builtin_prefetch(Bt + (size_t)(n0 + (lane & 15)) * Kdim + k + 32, 0, 3);
      __builtin_prefetch(Bt + (size_t)(n0 + 32 + (lane & 15)) * Kdim + k + 32, 0, 3);
    }
    v16h a0 = load_a_frag(A, Kdim, m0,      k, lane);
    v16h a1 = load_a_frag(A, Kdim, m0 + 16, k, lane);
#pragma unroll
    for (int ni = 0; ni < 4; ++ni) {
      v16h bf = load_b_frag(Bt, Kdim, n0 + ni * 16, k, lane);
      acc[0][ni] = __builtin_amdgcn_wmma_f32_16x16x32_f16(false, a0, false, bf, (short)0, acc[0][ni], false, false);
      acc[1][ni] = __builtin_amdgcn_wmma_f32_16x16x32_f16(false, a1, false, bf, (short)0, acc[1][ni], false, false);
    }
  }

  const int hi8 = (lane & 16) ? 8 : 0;
#pragma unroll
  for (int mi = 0; mi < 2; ++mi) {
#pragma unroll
    for (int ni = 0; ni < 4; ++ni) {
      const int gn = n0 + ni * 16 + (lane & 15);
      const float bv = bias[gn];
      const int rbase = m0 + mi * 16 + hi8;
#pragma unroll
      for (int r = 0; r < 8; ++r) {
        float v = acc[mi][ni][r] + bv;
        const size_t idx = (size_t)(rbase + r) * Ndim + gn;
        if constexpr (EPI == 0) {
          outh[idx] = (_Float16)v;
        } else if constexpr (EPI == 1) {
          outh[idx] = (_Float16)(0.5f * v * (1.0f + erff(v * 0.70710678118654752f)));
        } else {
          outf[idx] = v + resid[idx];
        }
      }
    }
  }
}

// ---------------- flash attention (causal, online softmax), 1 wave / 16-row q-tile ----------------
__global__ __launch_bounds__(32) void flash_attn(const _Float16* __restrict__ qkv,
                                                 const _Float16* __restrict__ Vt,
                                                 _Float16* __restrict__ outh) {
  constexpr int S = 2048, LD = 3072, NH = 16, Hh = 1024;
  const int lane = threadIdx.x;
  const int b = blockIdx.x / NH, h = blockIdx.x % NH;
  const int qbase = blockIdx.y * 16;
  const _Float16* base = qkv + (size_t)b * S * LD;
  const _Float16* Qh = base + h * 64;
  const _Float16* Kh = base + 1024 + h * 64;
  const _Float16* Vth = Vt + (size_t)blockIdx.x * 64 * S;  // [d][s], K-contiguous

  __shared__ __align__(16) _Float16 pbuf[16 * 32];

  const v16h aQ0 = load_a_frag(Qh, LD, qbase, 0,  lane);
  const v16h aQ1 = load_a_frag(Qh, LD, qbase, 32, lane);

  const int ncol = lane & 15;
  const int hi8  = (lane & 16) ? 8 : 0;

  float mrow[8], lrow[8];
  v8f o0 = {}, o1 = {}, o2 = {}, o3 = {};
#pragma unroll
  for (int r = 0; r < 8; ++r) { mrow[r] = -3.0e38f; lrow[r] = 0.0f; }

  const int jmax = (qbase + 47) >> 5;  // KV tiles of 32 covering rows <= qbase+15
  for (int j = 0; j < jmax; ++j) {
    const int kvb = j * 32;
    const v8f z = {};
    // S = Q * K^T  (reduction over d=64 -> two WMMAs per 16x16 score tile)
    v16h bK00 = load_b_frag(Kh, LD, kvb,      0,  lane);
    v16h bK01 = load_b_frag(Kh, LD, kvb,      32, lane);
    v16h bK10 = load_b_frag(Kh, LD, kvb + 16, 0,  lane);
    v16h bK11 = load_b_frag(Kh, LD, kvb + 16, 32, lane);
    v8f s0 = __builtin_amdgcn_wmma_f32_16x16x32_f16(false, aQ0, false, bK00, (short)0, z,  false, false);
    s0     = __builtin_amdgcn_wmma_f32_16x16x32_f16(false, aQ1, false, bK01, (short)0, s0, false, false);
    v8f s1 = __builtin_amdgcn_wmma_f32_16x16x32_f16(false, aQ0, false, bK10, (short)0, z,  false, false);
    s1     = __builtin_amdgcn_wmma_f32_16x16x32_f16(false, aQ1, false, bK11, (short)0, s1, false, false);

    float mnew[8], alpha[8];
#pragma unroll
    for (int r = 0; r < 8; ++r) {
      const int qi = qbase + r + hi8;
      float v0 = s0[r] * 0.125f;           // 1/sqrt(64)
      float v1 = s1[r] * 0.125f;
      if (kvb + ncol > qi)      v0 = -3.0e38f;
      if (kvb + 16 + ncol > qi) v1 = -3.0e38f;
      s0[r] = v0; s1[r] = v1;
      float t = fmaxf(v0, v1);             // row max across 16 lanes of the half-wave
      t = fmaxf(t, __shfl_xor(t, 1, 32));
      t = fmaxf(t, __shfl_xor(t, 2, 32));
      t = fmaxf(t, __shfl_xor(t, 4, 32));
      t = fmaxf(t, __shfl_xor(t, 8, 32));
      mnew[r]  = fmaxf(mrow[r], t);
      alpha[r] = __expf(mrow[r] - mnew[r]);
      mrow[r]  = mnew[r];
    }
#pragma unroll
    for (int r = 0; r < 8; ++r) {
      const float p0 = __expf(s0[r] - mnew[r]);
      const float p1 = __expf(s1[r] - mnew[r]);
      float rsum = p0 + p1;
      rsum += __shfl_xor(rsum, 1, 32);
      rsum += __shfl_xor(rsum, 2, 32);
      rsum += __shfl_xor(rsum, 4, 32);
      rsum += __shfl_xor(rsum, 8, 32);
      lrow[r] = lrow[r] * alpha[r] + rsum;
      pbuf[(r + hi8) * 32 + ncol]      = (_Float16)p0;   // C-layout -> LDS
      pbuf[(r + hi8) * 32 + 16 + ncol] = (_Float16)p1;
    }
#pragma unroll
    for (int r = 0; r < 8; ++r) { o0[r] *= alpha[r]; o1[r] *= alpha[r]; o2[r] *= alpha[r]; o3[r] *= alpha[r]; }
    __syncthreads();
    // reload P in A-fragment layout
    V16H ap;
    {
      const int rr = lane & 15;
      const int kk = (lane & 16) ? 8 : 0;
      ap.h[0] = *(const v8h*)&pbuf[rr * 32 + kk];
      ap.h[1] = *(const v8h*)&pbuf[rr * 32 + kk + 16];
    }
    // O += P * V : B-fragment of Vt is K-contiguous (two b128 loads per fragment)
#pragma unroll
    for (int dt = 0; dt < 4; ++dt) {
      v16h bv = load_b_frag(Vth, S, dt * 16, kvb, lane);
      v8f& o = (dt == 0) ? o0 : (dt == 1) ? o1 : (dt == 2) ? o2 : o3;
      o = __builtin_amdgcn_wmma_f32_16x16x32_f16(false, ap.v, false, bv, (short)0, o, false, false);
    }
    __syncthreads();
  }

#pragma unroll
  for (int dt = 0; dt < 4; ++dt) {
    const v8f& o = (dt == 0) ? o0 : (dt == 1) ? o1 : (dt == 2) ? o2 : o3;
#pragma unroll
    for (int r = 0; r < 8; ++r) {
      const size_t row = (size_t)b * S + qbase + r + hi8;
      outh[row * Hh + h * 64 + dt * 16 + ncol] = (_Float16)(o[r] / lrow[r]);
    }
  }
}

// ---------------- host-side orchestration ----------------
extern "C" void kernel_launch(void* const* d_in, const int* in_sizes, int n_in,
                              void* d_out, int out_size, void* d_ws, size_t ws_size,
                              hipStream_t stream) {
  (void)in_sizes; (void)n_in; (void)out_size; (void)ws_size;
  const float* x     = (const float*)d_in[0];
  const float* ln1w  = (const float*)d_in[1];
  const float* ln1b  = (const float*)d_in[2];
  const float* Wqkv  = (const float*)d_in[3];
  const float* bqkv  = (const float*)d_in[4];
  const float* Wo    = (const float*)d_in[5];
  const float* bo    = (const float*)d_in[6];
  const float* ln2w  = (const float*)d_in[7];
  const float* ln2b  = (const float*)d_in[8];
  const float* Wfc   = (const float*)d_in[9];
  const float* bfc   = (const float*)d_in[10];
  const float* Wproj = (const float*)d_in[11];
  const float* bproj = (const float*)d_in[12];
  float* out = (float*)d_out;

  // workspace layout (bytes); fc activation aliases the dead qkv+attn region
  char* ws = (char*)d_ws;
  _Float16* WqkvT  = (_Float16*)(ws + 0);               // 3072x1024 f16 = 6 MB
  _Float16* WoT    = (_Float16*)(ws + 6291456);         // 1024x1024 f16 = 2 MB
  _Float16* WfcT   = (_Float16*)(ws + 8388608);         // 4096x1024 f16 = 8 MB
  _Float16* WprojT = (_Float16*)(ws + 16777216);        // 1024x4096 f16 = 8 MB
  _Float16* lnbuf  = (_Float16*)(ws + 25165824);        // 4096x1024 f16 = 8 MB (reused ln1/ln2)
  _Float16* qkvbuf = (_Float16*)(ws + 33554432);        // 4096x3072 f16 = 24 MB
  _Float16* attnbuf= (_Float16*)(ws + 58720256);        // 4096x1024 f16 = 8 MB
  _Float16* fcbuf  = (_Float16*)(ws + 33554432);        // 4096x4096 f16 = 32 MB (alias)
  _Float16* vtbuf  = (_Float16*)(ws + 67108864);        // 32 heads x 64 x 2048 f16 = 8 MB

  // 1. weights -> f16, transposed to [N][K]
  convert_transpose<<<(3072 * 1024) / 256, 256, 0, stream>>>(Wqkv,  WqkvT,  1024, 3072);
  convert_transpose<<<(1024 * 1024) / 256, 256, 0, stream>>>(Wo,    WoT,    1024, 1024);
  convert_transpose<<<(4096 * 1024) / 256, 256, 0, stream>>>(Wfc,   WfcT,   1024, 4096);
  convert_transpose<<<(4096 * 1024) / 256, 256, 0, stream>>>(Wproj, WprojT, 4096, 1024);

  // 2. ln1(x) -> f16
  layernorm_f16<<<4096, 256, 0, stream>>>(x, ln1w, ln1b, lnbuf, 1024);
  // 3. qkv = ln1 @ Wqkv + b  (f16 out)
  gemm_wmma<0><<<dim3(3072 / 128, 4096 / 64), 128, 0, stream>>>(lnbuf, WqkvT, bqkv, nullptr, qkvbuf, nullptr, 3072, 1024);
  // 4. per-head V transpose for K-contiguous P*V fragments
  transpose_v<<<(2 * 16 * 64 * 2048) / 256, 256, 0, stream>>>(qkvbuf, vtbuf);
  // 5. causal flash attention
  flash_attn<<<dim3(32, 128), 32, 0, stream>>>(qkvbuf, vtbuf, attnbuf);
  // 6. x1 = x + attn @ Wo + b_o  (f32, into d_out)
  gemm_wmma<2><<<dim3(1024 / 128, 4096 / 64), 128, 0, stream>>>(attnbuf, WoT, bo, x, nullptr, out, 1024, 1024);
  // 7. ln2(x1) -> f16
  layernorm_f16<<<4096, 256, 0, stream>>>(out, ln2w, ln2b, lnbuf, 1024);
  // 8. fc = gelu(ln2 @ Wfc + b_fc)  (f16 out)
  gemm_wmma<1><<<dim3(4096 / 128, 4096 / 64), 128, 0, stream>>>(lnbuf, WfcT, bfc, nullptr, fcbuf, nullptr, 4096, 1024);
  // 9. out = x1 + fc @ Wproj + b_proj  (in-place residual from d_out)
  gemm_wmma<2><<<dim3(1024 / 128, 4096 / 64), 128, 0, stream>>>(fcbuf, WprojT, bproj, out, nullptr, out, 1024, 4096);
}
